// AttentionAllocationSystem_54966991454361
// MI455X (gfx1250) — compile-verified, matrix operand order checked by reading, not run on hardware
//
#include <hip/hip_runtime.h>

typedef __attribute__((ext_vector_type(16))) _Float16 v16h;
typedef __attribute__((ext_vector_type(8)))  _Float16 v8h;
typedef __attribute__((ext_vector_type(8)))  float    v8f;

#define NB   64
#define ND   128
#define NA   128
#define DEFD 6
#define ATKD 6
#define CTXD 16

// ---------------------------------------------------------------------------
// Kernel 1: QKV net + attention + layer-1 factorization.
//   U[b,d,0:64] = enriched(d) @ eW1[0:22,:] + eb1
//   V[b,a,0:64] = attacker(a) @ eW1[22:28,:]
// One block per batch, thread t = defender index = attacker index.
// ---------------------------------------------------------------------------
__global__ __launch_bounds__(128) void aas_prep_kernel(
    const float* __restrict__ defF, const float* __restrict__ atkF,
    const float* __restrict__ fW1, const float* __restrict__ fb1,
    const float* __restrict__ fW2, const float* __restrict__ fb2,
    const float* __restrict__ qW,  const float* __restrict__ qb,
    const float* __restrict__ kW,  const float* __restrict__ kb,
    const float* __restrict__ vW,  const float* __restrict__ vb,
    const float* __restrict__ eW1, const float* __restrict__ eb1,
    float* __restrict__ U, float* __restrict__ V)
{
    __shared__ float Ks[ND][CTXD];
    __shared__ float Vs[ND][CTXD];

    const int b = blockIdx.x;
    const int t = threadIdx.x;

    // ---- defender feature net -> h2[32] -> q (regs), k/v (LDS) ----
    float df[DEFD];
    {
        const float* dp = defF + (b * ND + t) * DEFD;
        #pragma unroll
        for (int i = 0; i < DEFD; ++i) df[i] = dp[i];
    }
    float h1[32];
    #pragma unroll
    for (int j = 0; j < 32; ++j) {
        float s = fb1[j];
        #pragma unroll
        for (int i = 0; i < DEFD; ++i) s += df[i] * fW1[i * 32 + j];
        h1[j] = fmaxf(s, 0.0f);
    }
    float h2[32];
    #pragma unroll
    for (int j = 0; j < 32; ++j) {
        float s = fb2[j];
        #pragma unroll
        for (int kk = 0; kk < 32; ++kk) s += h1[kk] * fW2[kk * 32 + j];
        h2[j] = fmaxf(s, 0.0f);
    }
    float q[CTXD];
    #pragma unroll
    for (int c = 0; c < CTXD; ++c) {
        float sq = qb[c], sk = kb[c], sv = vb[c];
        #pragma unroll
        for (int kk = 0; kk < 32; ++kk) {
            const float h = h2[kk];
            sq += h * qW[kk * CTXD + c];
            sk += h * kW[kk * CTXD + c];
            sv += h * vW[kk * CTXD + c];
        }
        q[c] = sq;
        Ks[t][c] = sk;
        Vs[t][c] = sv;
    }
    __syncthreads();

    // ---- online softmax attention over the 128 defenders ----
    float m = -3.0e38f, l = 0.0f;
    float ctx[CTXD];
    #pragma unroll
    for (int c = 0; c < CTXD; ++c) ctx[c] = 0.0f;
    for (int e = 0; e < ND; ++e) {
        float s = 0.0f;
        #pragma unroll
        for (int c = 0; c < CTXD; ++c) s += q[c] * Ks[e][c];
        s *= 0.25f;                                  // 1/sqrt(16)
        const float nm   = fmaxf(s, m);
        const float corr = __expf(m - nm);
        const float p    = __expf(s - nm);
        l = l * corr + p;
        #pragma unroll
        for (int c = 0; c < CTXD; ++c) ctx[c] = ctx[c] * corr + p * Vs[e][c];
        m = nm;
    }
    const float inv_l = 1.0f / l;
    #pragma unroll
    for (int c = 0; c < CTXD; ++c) ctx[c] *= inv_l;

    // ---- U row: enriched(22) @ eW1[0:22,:] + eb1 ----
    float* Urow = U + (b * ND + t) * 64;
    #pragma unroll 8
    for (int kcol = 0; kcol < 64; ++kcol) {
        float s = eb1[kcol];
        #pragma unroll
        for (int i = 0; i < DEFD; ++i) s += df[i] * eW1[i * 64 + kcol];
        #pragma unroll
        for (int c = 0; c < CTXD; ++c) s += ctx[c] * eW1[(DEFD + c) * 64 + kcol];
        Urow[kcol] = s;
    }

    // ---- V row: attacker(6) @ eW1[22:28,:] ----
    float af[ATKD];
    {
        const float* ap = atkF + (b * NA + t) * ATKD;
        #pragma unroll
        for (int i = 0; i < ATKD; ++i) af[i] = ap[i];
    }
    float* Vrow = V + (b * NA + t) * 64;
    #pragma unroll 8
    for (int kcol = 0; kcol < 64; ++kcol) {
        float s = 0.0f;
        #pragma unroll
        for (int i = 0; i < ATKD; ++i) s += af[i] * eW1[(22 + i) * 64 + kcol];
        Vrow[kcol] = s;
    }
}

// ---------------------------------------------------------------------------
// B-fragment loader: W row-major [K x ncols]; lane = column (lane&15),
// half-select (lane>>4) picks K sub-blocks, mirroring the 16-bit A layout.
// ---------------------------------------------------------------------------
__device__ __forceinline__ v16h load_bfrag(const float* __restrict__ W,
                                           int ncols, int col, int kbase, int hs)
{
    v16h r;
    #pragma unroll
    for (int g = 0; g < 2; ++g) {
        #pragma unroll
        for (int i = 0; i < 8; ++i) {
            const int kk = kbase + g * 16 + hs * 8 + i;
            r[g * 8 + i] = (_Float16)W[kk * ncols + col];
        }
    }
    return r;
}

// ---------------------------------------------------------------------------
// Kernel 2: pairwise MLP on the matrix pipe.
// Wave owns one (b, d); loops over 8 attacker tiles of 16 pairs.
// Per tile: layer2 = 8x v_wmma_f32_16x16x32_f16, layer3 = 4x, head = VALU dot.
// LDS transpose buffers are ping-ponged on (at & 1) so consecutive tiles are
// alias-free and the scheduler can pipeline across the C->A transpose.
// ---------------------------------------------------------------------------
__global__ __launch_bounds__(256) void aas_pair_kernel(
    const float* __restrict__ U,  const float* __restrict__ V,
    const float* __restrict__ eW2, const float* __restrict__ eb2,
    const float* __restrict__ eW3, const float* __restrict__ eb3,
    const float* __restrict__ eW4, const float* __restrict__ eb4,
    float* __restrict__ out)
{
    __shared__ __align__(16) _Float16 x2s[2][8][16][64];  // layer2 out (ping-pong)
    __shared__ __align__(16) float    x3s[2][8][16][32];  // layer3 out (ping-pong)
    __shared__ float w4s[32];

    const int lane = threadIdx.x & 31;
    const int wave = threadIdx.x >> 5;
    const int gw   = blockIdx.x * 8 + wave;   // gw = b*128 + d
    const int b    = gw >> 7;

    if (threadIdx.x < 32) w4s[threadIdx.x] = eW4[threadIdx.x];
    __syncthreads();

    const int nrow = lane & 15;   // WMMA row (A) / column (B,C)
    const int hs   = lane >> 4;   // half-select for K sub-blocks
    const int rbase = hs * 8;
    const float bias4 = eb4[0];

    // ---- stage all weight B-fragments once per wave (reused over 8 tiles) ----
    v16h B2f[4][2];
    #pragma unroll
    for (int nt = 0; nt < 4; ++nt)
        #pragma unroll
        for (int f = 0; f < 2; ++f)
            B2f[nt][f] = load_bfrag(eW2, 64, nt * 16 + nrow, f * 32, hs);
    v16h B3f[2][2];
    #pragma unroll
    for (int nt = 0; nt < 2; ++nt)
        #pragma unroll
        for (int f = 0; f < 2; ++f)
            B3f[nt][f] = load_bfrag(eW3, 32, nt * 16 + nrow, f * 32, hs);

    // ---- per-lane slice of U[d] (the K offsets this lane owns) ----
    const float* Urow = U + (gw << 6);
    float ub[4][8];
    #pragma unroll
    for (int f = 0; f < 2; ++f)
        #pragma unroll
        for (int g = 0; g < 2; ++g) {
            const int base = f * 32 + g * 16 + hs * 8;
            const float4 u0 = *(const float4*)(Urow + base);
            const float4 u1 = *(const float4*)(Urow + base + 4);
            ub[f * 2 + g][0] = u0.x; ub[f * 2 + g][1] = u0.y;
            ub[f * 2 + g][2] = u0.z; ub[f * 2 + g][3] = u0.w;
            ub[f * 2 + g][4] = u1.x; ub[f * 2 + g][5] = u1.y;
            ub[f * 2 + g][6] = u1.z; ub[f * 2 + g][7] = u1.w;
        }

    #pragma unroll 2
    for (int at = 0; at < 8; ++at) {
        const int pp = at & 1;   // ping-pong LDS buffer select

        // ---- layer1: A = relu(U[d] + V[a]), 16-bit A layout, K=64 as 2 frags ----
        const float* Vrow = V + ((b * NA + at * 16 + nrow) << 6);
        v16h A[2];
        #pragma unroll
        for (int f = 0; f < 2; ++f)
            #pragma unroll
            for (int g = 0; g < 2; ++g) {
                const int base = f * 32 + g * 16 + hs * 8;
                const float4 v0 = *(const float4*)(Vrow + base);
                const float4 v1 = *(const float4*)(Vrow + base + 4);
                float xx[8] = { v0.x, v0.y, v0.z, v0.w, v1.x, v1.y, v1.z, v1.w };
                #pragma unroll
                for (int i = 0; i < 8; ++i) {
                    const float x = fmaxf(xx[i] + ub[f * 2 + g][i], 0.0f);
                    A[f][g * 8 + i] = (_Float16)x;
                }
            }

        // ---- layer2: [16x64] @ [64x64], bias pre-loaded into accumulator ----
        #pragma unroll
        for (int nt = 0; nt < 4; ++nt) {
            v8f c;
            const float bias = eb2[nt * 16 + nrow];
            #pragma unroll
            for (int j = 0; j < 8; ++j) c[j] = bias;
            c = __builtin_amdgcn_wmma_f32_16x16x32_f16(false, A[0], false, B2f[nt][0],
                                                       (short)0, c, false, false);
            c = __builtin_amdgcn_wmma_f32_16x16x32_f16(false, A[1], false, B2f[nt][1],
                                                       (short)0, c, false, false);
            const int col = nt * 16 + nrow;
            #pragma unroll
            for (int j = 0; j < 8; ++j) {
                const float x = fmaxf(c[j], 0.0f);
                x2s[pp][wave][rbase + j][col] = (_Float16)x;  // C layout -> transpose
            }
        }

        // ---- rebuild A fragments for layer3 from per-wave LDS (in-order DS) ----
        v16h A3[2];
        const _Float16* rp = &x2s[pp][wave][nrow][0];
        #pragma unroll
        for (int f = 0; f < 2; ++f)
            #pragma unroll
            for (int g = 0; g < 2; ++g) {
                const v8h ch = *(const v8h*)(rp + f * 32 + g * 16 + hs * 8);
                #pragma unroll
                for (int i = 0; i < 8; ++i) A3[f][g * 8 + i] = ch[i];
            }

        // ---- layer3: [16x64] @ [64x32] ----
        #pragma unroll
        for (int nt = 0; nt < 2; ++nt) {
            v8f c;
            const float bias = eb3[nt * 16 + nrow];
            #pragma unroll
            for (int j = 0; j < 8; ++j) c[j] = bias;
            c = __builtin_amdgcn_wmma_f32_16x16x32_f16(false, A3[0], false, B3f[nt][0],
                                                       (short)0, c, false, false);
            c = __builtin_amdgcn_wmma_f32_16x16x32_f16(false, A3[1], false, B3f[nt][1],
                                                       (short)0, c, false, false);
            const int col = nt * 16 + nrow;
            #pragma unroll
            for (int j = 0; j < 8; ++j)
                x3s[pp][wave][rbase + j][col] = fmaxf(c[j], 0.0f);
        }

        // ---- head: [16x32] @ [32x1]; lane n owns output row n ----
        if (lane < 16) {
            float acc = bias4;
            const float* xr = &x3s[pp][wave][lane][0];
            #pragma unroll
            for (int kk = 0; kk < 8; ++kk) {
                const float4 x4 = *(const float4*)(xr + kk * 4);
                acc += x4.x * w4s[kk * 4 + 0];
                acc += x4.y * w4s[kk * 4 + 1];
                acc += x4.z * w4s[kk * 4 + 2];
                acc += x4.w * w4s[kk * 4 + 3];
            }
            out[(gw << 7) + at * 16 + lane] = acc;
        }
    }
}

extern "C" void kernel_launch(void* const* d_in, const int* in_sizes, int n_in,
                              void* d_out, int out_size, void* d_ws, size_t ws_size,
                              hipStream_t stream) {
    (void)in_sizes; (void)n_in; (void)out_size; (void)ws_size;
    const float* defF = (const float*)d_in[0];
    const float* atkF = (const float*)d_in[1];
    const float* fW1  = (const float*)d_in[2];
    const float* fb1  = (const float*)d_in[3];
    const float* fW2  = (const float*)d_in[4];
    const float* fb2  = (const float*)d_in[5];
    const float* qW   = (const float*)d_in[6];
    const float* qb   = (const float*)d_in[7];
    const float* kW   = (const float*)d_in[8];
    const float* kb   = (const float*)d_in[9];
    const float* vW   = (const float*)d_in[10];
    const float* vb   = (const float*)d_in[11];
    const float* eW1  = (const float*)d_in[12];
    const float* eb1  = (const float*)d_in[13];
    const float* eW2  = (const float*)d_in[14];
    const float* eb2  = (const float*)d_in[15];
    const float* eW3  = (const float*)d_in[16];
    const float* eb3  = (const float*)d_in[17];
    const float* eW4  = (const float*)d_in[18];
    const float* eb4  = (const float*)d_in[19];
    float* out = (float*)d_out;

    float* U  = (float*)d_ws;                // [64*128, 64] f32 = 2 MB
    float* Vw = U + NB * ND * 64;            // [64*128, 64] f32 = 2 MB

    aas_prep_kernel<<<NB, ND, 0, stream>>>(defF, atkF, fW1, fb1, fW2, fb2,
                                           qW, qb, kW, kb, vW, vb, eW1, eb1, U, Vw);
    // 8192 waves, 8 waves per 256-thread block
    aas_pair_kernel<<<(NB * ND) / 8, 256, 0, stream>>>(U, Vw, eW2, eb2, eW3, eb3,
                                                       eW4, eb4, out);
}
